// Decoder_37074157699394
// MI455X (gfx1250) — matrix operand order repaired and checked by reading, not compiled, not run on hardware
//
#include <hip/hip_runtime.h>
#include <stdint.h>

// ---------------------------------------------------------------------------
// 2-layer LSTM decoder (H=100), T=8192 teacher-forced + F=1024 autoregressive
// steps. Single persistent workgroup (416 threads = 13 wave32) on one WGP.
// Weights live in LDS as f16 in WMMA-B register-image layout. Matvecs use
// V_WMMA_F32_16X16X32_F16 for K=0..95 (3 chunks); the K=96..99 remainder is
// folded into gate extraction as 4 f32 FMAs per gate from a small LDS table
// (kills the 7/8-zero padding chunk -> ~17% less LDS traffic per step).
// B tiles for each layer are batch-loaded into registers before the WMMA
// chain so ds_load_b128s issue back-to-back.
// ---------------------------------------------------------------------------

#define H        100
#define HP       128
#define KWM      96       // K handled by WMMA (3 chunks of 32)
#define G4       400      // 4*H gate rows
#define NT       25       // 400/16 N-tiles
#define KC       3        // WMMA K chunks
#define TSEQ     8192
#define TFUT     1024
#define NMAT     3        // w_hh1, w_ih2, w_hh2
#define BLK_W    256      // dwords per B tile-chunk (32 lanes * 8 dwords)
#define WWORDS   (NMAT * NT * KC * BLK_W)   // 57600 dwords = 225 KB
#define RWORDS   (NMAT * G4 * 4)            // 4800 f32 remainder weights
#define NTHREADS 416      // 13 wave32: 2 N-tiles per wave (wave 12 clamps)

typedef __attribute__((ext_vector_type(16))) _Float16 v16h;
typedef __attribute__((ext_vector_type(8)))  float    v8f;

__device__ __forceinline__ int blk_idx(int m, int tile, int kc) {
    return ((m * NT + tile) * KC + kc);
}

__device__ __forceinline__ float sigm(float x) {
    return 1.0f / (1.0f + __expf(-x));
}
__device__ __forceinline__ float tanh_f(float x) {
    return 2.0f * sigm(2.0f * x) - 1.0f;   // == tanh(x)
}

__device__ __forceinline__ float dot4(float4 a, float4 b) {
    return a.x * b.x + a.y * b.y + a.z * b.z + a.w * b.w;
}

__device__ __forceinline__ uint32_t pack2h(float lo, float hi) {
    union { _Float16 h[2]; uint32_t u; } p;
    p.h[0] = (_Float16)lo;
    p.h[1] = (_Float16)hi;
    return p.u;
}

// Load one 32x16 f16 B tile-chunk from LDS (per-lane 32B register image).
__device__ __forceinline__ v16h load_b(const uint32_t* p, int kc) {
    union { uint4 q[2]; v16h h; } r;
    const uint4* q = (const uint4*)(p + kc * BLK_W);
    r.q[0] = q[0];
    r.q[1] = q[1];
    return r.h;
}

// Build A (16x32 f16) with the h vector in row M=0.  Rows 1..15 get copies of
// the same data (harmless: only row 0 of C is extracted).  Per the ISA A
// layout, each lane's 8 dwords are two contiguous 4-dword runs of h-pairs:
//   lanes 0-15 : K = kc*32 + {0..7}  and  kc*32 + {16..23}
//   lanes16-31 : K = kc*32 + {8..15} and  kc*32 + {24..31}
// hptr is (uint32*)hbuf pre-offset by +4 dwords for lanes >= 16.
__device__ __forceinline__ v16h build_a(const uint32_t* hptr, int kc) {
    union { uint4 q[2]; v16h h; } r;
    r.q[0] = *(const uint4*)(hptr + kc * 16);
    r.q[1] = *(const uint4*)(hptr + kc * 16 + 8);
    return r.h;
}

__device__ __forceinline__ v8f wmma_f16(v16h a, v16h b, v8f c) {
    return __builtin_amdgcn_wmma_f32_16x16x32_f16(
        false, a, false, b, (short)0, c, false, false);
}

__global__ void __launch_bounds__(NTHREADS, 1)
decoder_lstm_kernel(const float* __restrict__ input,
                    const float* __restrict__ enc_h,
                    const float* __restrict__ enc_c,
                    const float* __restrict__ w_ih1,   // (400,1)
                    const float* __restrict__ w_hh1,   // (400,100)
                    const float* __restrict__ b_ih1,
                    const float* __restrict__ b_hh1,
                    const float* __restrict__ w_ih2,   // (400,100)
                    const float* __restrict__ w_hh2,   // (400,100)
                    const float* __restrict__ b_ih2,
                    const float* __restrict__ b_hh2,
                    const float* __restrict__ w_lin,   // (1,100)
                    const float* __restrict__ b_lin,   // (1,)
                    float* __restrict__ out)           // (9216,)
{
    extern __shared__ uint32_t wlds[];   // 225 KB f16 B tiles + 19 KB f32 rem
    float* remw = (float*)(wlds + WWORDS);

    __shared__ float    Gs[G4];
    __shared__ float    bias1[G4], bias2[G4], wih1c[G4];
    __shared__ float    wlin_s[HP], c1s[HP], c2s[HP];
    __shared__ __align__(16) float    h1f[HP];
    __shared__ __align__(16) float    h2f[HP];
    __shared__ __align__(16) _Float16 h1h[HP];
    __shared__ __align__(16) _Float16 h2h[HP];
    __shared__ float    ybuf;

    const int tid  = threadIdx.x;
    const int lane = tid & 31;
    const int wave = tid >> 5;

    // ---- one-time setup: pack weights to f16 WMMA-B layout in LDS ----
    for (int idx = tid; idx < WWORDS; idx += NTHREADS) {
        int blk  = idx >> 8;
        int win  = idx & 255;
        int l    = win >> 3;
        int v    = win & 7;
        int m    = blk / (NT * KC);
        int rem  = blk % (NT * KC);
        int tile = rem / KC;
        int kc   = rem % KC;
        int n    = tile * 16 + (l & 15);
        int k0   = kc * 32 + 8 * (l >> 4) + 2 * v + ((v >= 4) ? 8 : 0); // < 96 < H
        const float* W = (m == 0) ? w_hh1 : ((m == 1) ? w_ih2 : w_hh2);
        wlds[idx] = pack2h(W[n * H + k0], W[n * H + k0 + 1]);
    }
    // K=96..99 remainder weights, f32: remw[(m*400+n)*4 + j] = W[n][96+j]
    for (int i = tid; i < NMAT * G4; i += NTHREADS) {
        int m = i / G4;
        int n = i % G4;
        const float* W = (m == 0) ? w_hh1 : ((m == 1) ? w_ih2 : w_hh2);
        remw[i * 4 + 0] = W[n * H + 96];
        remw[i * 4 + 1] = W[n * H + 97];
        remw[i * 4 + 2] = W[n * H + 98];
        remw[i * 2 * 2 + 3] = W[n * H + 99];
    }
    for (int i = tid; i < G4; i += NTHREADS) {
        bias1[i] = b_ih1[i] + b_hh1[i];
        bias2[i] = b_ih2[i] + b_hh2[i];
        wih1c[i] = w_ih1[i];
    }
    if (tid < HP) {
        wlin_s[tid] = (tid < H) ? w_lin[tid] : 0.0f;
        c1s[tid]    = (tid < H) ? enc_c[tid] : 0.0f;
        h1f[tid]    = (tid < H) ? enc_h[tid] : 0.0f;
        h1h[tid]    = (_Float16)h1f[tid];
        c2s[tid]    = 0.0f;
        h2f[tid]    = 0.0f;
        h2h[tid]    = (_Float16)0.0f;
    }
    if (tid == 0) ybuf = 0.0f;
    const float blin = b_lin[0];

    // Branchless tile assignment: wave w owns tiles 2w, 2w+1; wave 12's
    // second tile clamps to 24 (recomputes tile 24, stores identical data).
    int t0 = 2 * wave;
    int t1 = 2 * wave + 1;
    if (t1 >= NT) t1 = NT - 1;
    if (t0 >= NT) t0 = NT - 1;
    const int n0 = t0 * 16 + lane;     // gate row for this lane (lane<16)
    const int n1 = t1 * 16 + lane;

    // Per-lane base pointers (hoisted out of the time loop).
    const int laneoff = lane * 8;
    const uint32_t* pB1a = wlds + blk_idx(0, t0, 0) * BLK_W + laneoff;
    const uint32_t* pB1b = wlds + blk_idx(0, t1, 0) * BLK_W + laneoff;
    const uint32_t* pB2a = wlds + blk_idx(1, t0, 0) * BLK_W + laneoff;
    const uint32_t* pB2b = wlds + blk_idx(1, t1, 0) * BLK_W + laneoff;
    const uint32_t* pB3a = wlds + blk_idx(2, t0, 0) * BLK_W + laneoff;
    const uint32_t* pB3b = wlds + blk_idx(2, t1, 0) * BLK_W + laneoff;
    const float4* remw4 = (const float4*)remw;
    const int hoff = (lane >= 16) ? 4 : 0;
    const uint32_t* pH1 = (const uint32_t*)h1h + hoff;
    const uint32_t* pH2 = (const uint32_t*)h2h + hoff;

    __syncthreads();

    float xpre = input[0];     // software-pipelined x prefetch

#pragma unroll 1
    for (int t = 0; t < TSEQ + TFUT; ++t) {
        const float x = (t < TSEQ) ? xpre : ybuf;
        if (t + 1 < TSEQ) xpre = input[t + 1];     // overlaps this whole step

        // ---------- layer 1: gates1 = w_hh1 @ h1 + x*w_ih1 + bias ----------
        v16h a1[KC], bA[KC], bB[KC];
#pragma unroll
        for (int kc = 0; kc < KC; ++kc) {          // batch all loads first
            a1[kc] = build_a(pH1, kc);
            bA[kc] = load_b(pB1a, kc);
            bB[kc] = load_b(pB1b, kc);
        }
        v8f ca = {};
        v8f cb = {};
#pragma unroll
        for (int kc = 0; kc < KC; ++kc) {
            ca = wmma_f16(a1[kc], bA[kc], ca);
            cb = wmma_f16(a1[kc], bB[kc], cb);
        }
        if (lane < 16) {
            float4 hq = *(const float4*)(h1f + KWM);        // h1[96..99]
            float4 w0 = remw4[0 * G4 + n0];
            float4 w1 = remw4[0 * G4 + n1];
            Gs[n0] = ca[0] + x * wih1c[n0] + bias1[n0] + dot4(w0, hq);
            Gs[n1] = cb[0] + x * wih1c[n1] + bias1[n1] + dot4(w1, hq);
        }
        __syncthreads();

        // layer-1 elementwise update
        if (tid < H) {
            float gi = Gs[tid],       gf = Gs[100 + tid];
            float gg = Gs[200 + tid], go = Gs[300 + tid];
            float c  = sigm(gf) * c1s[tid] + sigm(gi) * tanh_f(gg);
            c1s[tid] = c;
            float h  = sigm(go) * tanh_f(c);
            h1f[tid] = h;
            h1h[tid] = (_Float16)h;
        }
        __syncthreads();

        // ---------- layer 2: gates2 = w_ih2 @ h1 + w_hh2 @ h2 + bias -------
        v16h a2[KC], bC[KC], bD[KC], bE[KC], bF[KC];
#pragma unroll
        for (int kc = 0; kc < KC; ++kc) {          // batch all loads first
            a1[kc] = build_a(pH1, kc);
            a2[kc] = build_a(pH2, kc);
            bC[kc] = load_b(pB2a, kc);
            bD[kc] = load_b(pB3a, kc);
            bE[kc] = load_b(pB2b, kc);
            bF[kc] = load_b(pB3b, kc);
        }
        ca = (v8f){};
        cb = (v8f){};
#pragma unroll
        for (int kc = 0; kc < KC; ++kc) {
            ca = wmma_f16(a1[kc], bC[kc], ca);
            ca = wmma_f16(a2[kc], bD[kc], ca);
            cb = wmma_f16(a1[kc], bE[kc], cb);
            cb = wmma_f16(a2[kc], bF[kc], cb);
        }
        if (lane < 16) {
            float4 hq1 = *(const float4*)(h1f + KWM);       // h1[96..99]
            float4 hq2 = *(const float4*)(h2f + KWM);       // h2[96..99]
            float4 wA0 = remw4[1 * G4 + n0];
            float4 wB0 = remw4[2 * G4 + n0];
            float4 wA1 = remw4[1 * G4 + n1];
            float4 wB1 = remw4[2 * G4 + n1];
            Gs[n0] = ca[0] + bias2[n0] + dot4(wA0, hq1) + dot4(wB0, hq2);
            Gs[n1] = cb[0] + bias2[n1] + dot4(wA1, hq1) + dot4(wB1, hq2);
        }
        __syncthreads();

        // layer-2 elementwise update
        if (tid < H) {
            float gi = Gs[tid],       gf = Gs[100 + tid];
            float gg = Gs[200 + tid], go = Gs[300 + tid];
            float c  = sigm(gf) * c2s[tid] + sigm(gi) * tanh_f(gg);
            c2s[tid] = c;
            float h  = sigm(go) * tanh_f(c);
            h2h[tid] = (_Float16)h;
            h2f[tid] = h;
        }
        __syncthreads();

        // ---------- output projection: y = h2 . w_lin + b_lin --------------
        if (wave == 0) {
            float s = h2f[lane]      * wlin_s[lane]
                    + h2f[lane + 32] * wlin_s[lane + 32]
                    + h2f[lane + 64] * wlin_s[lane + 64]
                    + h2f[lane + 96] * wlin_s[lane + 96];
#pragma unroll
            for (int off = 16; off > 0; off >>= 1)
                s += __shfl_xor(s, off, 32);
            if (lane == 0) {
                float y = s + blin;
                out[t]  = y;
                ybuf    = y;
            }
        }
        __syncthreads();
    }
}

extern "C" void kernel_launch(void* const* d_in, const int* in_sizes, int n_in,
                              void* d_out, int out_size, void* d_ws, size_t ws_size,
                              hipStream_t stream) {
    (void)in_sizes; (void)n_in; (void)out_size; (void)d_ws; (void)ws_size;
    const float* input  = (const float*)d_in[0];
    const float* enc_h  = (const float*)d_in[1];
    const float* enc_c  = (const float*)d_in[2];
    const float* w_ih1  = (const float*)d_in[3];
    const float* w_hh1  = (const float*)d_in[4];
    const float* b_ih1  = (const float*)d_in[5];
    const float* b_hh1  = (const float*)d_in[6];
    const float* w_ih2  = (const float*)d_in[7];
    const float* w_hh2  = (const float*)d_in[8];
    const float* b_ih2  = (const float*)d_in[9];
    const float* b_hh2  = (const float*)d_in[10];
    const float* w_lin  = (const float*)d_in[11];
    const float* b_lin  = (const float*)d_in[12];
    float* out = (float*)d_out;

    const size_t shmem = (size_t)(WWORDS + RWORDS) * sizeof(uint32_t); // ~244 KB
    decoder_lstm_kernel<<<dim3(1), dim3(NTHREADS), shmem, stream>>>(
        input, enc_h, enc_c,
        w_ih1, w_hh1, b_ih1, b_hh1,
        w_ih2, w_hh2, b_ih2, b_hh2,
        w_lin, b_lin, out);
}
